// Net_65025804862040
// MI455X (gfx1250) — matrix operand order repaired and checked by reading, not compiled, not run on hardware
//
#include <hip/hip_runtime.h>

typedef __attribute__((ext_vector_type(16))) _Float16 v16h;
typedef __attribute__((ext_vector_type(8)))  float    v8f;

// ---------------- degree / norm ----------------
__global__ void k_deg_init(float* __restrict__ deg, int N) {
  int i = blockIdx.x * blockDim.x + threadIdx.x;
  if (i < N) deg[i] = 1.0f;  // self-loop contributes 1 to deg[row]
}

__global__ void k_deg_count(const int* __restrict__ row, float* __restrict__ deg, int E) {
  int e = blockIdx.x * blockDim.x + threadIdx.x;
  if (e < E) unsafeAtomicAdd(&deg[row[e]], 1.0f);  // global_atomic_add_f32
}

__global__ void k_deg_rsqrt(float* __restrict__ dinv, int N) {
  int i = blockIdx.x * blockDim.x + threadIdx.x;
  if (i < N) dinv[i] = rsqrtf(dinv[i]);  // deg >= 1 guaranteed
}

// ---------------- GEMM1: h0 = x @ W1^T + b1 via v_wmma_f32_16x16x32_f16 ----
// One wave computes a full 16x32 output stripe (both 16-col tiles) so each
// A fragment (the x tile) is loaded from HBM exactly once. K=128 in four
// K=32 WMMA steps x 2 col tiles = 8 WMMAs per wave.
// A fragment as loaded here (contiguous k, not ISA table order): lane<16 -> M=lane, K=0..15;
// lane>=16 -> M=lane-16, K=16..31. B 32x16, same k order: lane<16 -> N=lane,K=0..15;
// lane>=16 -> N=lane-16, K=16..31. C/D: VGPR r -> M=r (+8 for hi lanes), N=lane%16.
__global__ __launch_bounds__(256) void k_gemm1_wmma(
    const float* __restrict__ x, const float* __restrict__ W1,
    const float* __restrict__ b1, float* __restrict__ h0, int N) {
  const int wave = blockIdx.x * (blockDim.x >> 5) + (threadIdx.x >> 5);
  const int lane = threadIdx.x & 31;
  const int nRowTiles = (N + 15) >> 4;
  if (wave >= nRowTiles) return;
  const int r0 = wave << 4;              // 16-row tile
  const int m  = lane & 15;
  const bool hi = lane >= 16;

  const int arow = min(r0 + m, N - 1);
  const float* aptr  = x  + (size_t)arow * 128;
  const float* bptr0 = W1 + (size_t)(m)      * 128;  // B col n (tile 0) = W1 row n
  const float* bptr1 = W1 + (size_t)(16 + m) * 128;  // B col n (tile 1)
  const int kA  = hi ? 16 : 0;
  const int kA2 = hi ? 24 : 8;
  const int kB  = hi ? 16 : 0;

  v8f c0 = {}, c1 = {};
#pragma unroll
  for (int k0 = 0; k0 < 128; k0 += 32) {
    float af[16], bf0[16], bf1[16];
    *(float4*)(af + 0)   = *(const float4*)(aptr + k0 + kA);
    *(float4*)(af + 4)   = *(const float4*)(aptr + k0 + kA + 4);
    *(float4*)(af + 8)   = *(const float4*)(aptr + k0 + kA2);
    *(float4*)(af + 12)  = *(const float4*)(aptr + k0 + kA2 + 4);
    *(float4*)(bf0 + 0)  = *(const float4*)(bptr0 + k0 + kB);
    *(float4*)(bf0 + 4)  = *(const float4*)(bptr0 + k0 + kB + 4);
    *(float4*)(bf0 + 8)  = *(const float4*)(bptr0 + k0 + kB + 8);
    *(float4*)(bf0 + 12) = *(const float4*)(bptr0 + k0 + kB + 12);
    *(float4*)(bf1 + 0)  = *(const float4*)(bptr1 + k0 + kB);
    *(float4*)(bf1 + 4)  = *(const float4*)(bptr1 + k0 + kB + 4);
    *(float4*)(bf1 + 8)  = *(const float4*)(bptr1 + k0 + kB + 8);
    *(float4*)(bf1 + 12) = *(const float4*)(bptr1 + k0 + kB + 12);
    v16h a, b0, b1v;
#pragma unroll
    for (int i = 0; i < 16; ++i) {
      a[i]   = (_Float16)af[i];
      b0[i]  = (_Float16)bf0[i];
      b1v[i] = (_Float16)bf1[i];
    }
    c0 = __builtin_amdgcn_wmma_f32_16x16x32_f16(false, a, false, b0,  (short)0, c0,
                                                false, false);
    c1 = __builtin_amdgcn_wmma_f32_16x16x32_f16(false, a, false, b1v, (short)0, c1,
                                                false, false);
  }
  const float bias0 = b1[m];
  const float bias1 = b1[16 + m];
  const int rb = r0 + (hi ? 8 : 0);
#pragma unroll
  for (int r = 0; r < 8; ++r) {
    const int rr = rb + r;
    if (rr < N) {
      h0[(size_t)rr * 32 + m]      = c0[r] + bias0;
      h0[(size_t)rr * 32 + 16 + m] = c1[r] + bias1;
    }
  }
}

// ------------- self-loop init: acc[i] = dinv[i]^2 * h[i]  (F feats) -------
template <int F>
__global__ void k_self_loop(const float* __restrict__ h, const float* __restrict__ dinv,
                            float* __restrict__ acc, int N) {
  constexpr int CH = F / 4;  // float4 chunks per node
  int tid = blockIdx.x * blockDim.x + threadIdx.x;
  int i = tid / CH;
  if (i >= N) return;
  int f = (tid % CH) * 4;
  float d = dinv[i];
  float s = d * d;                              // norm of self-loop = 1/deg
  float4 hv = *(const float4*)(h + (size_t)i * F + f);
  float4 r; r.x = s*hv.x; r.y = s*hv.y; r.z = s*hv.z; r.w = s*hv.w;
  *(float4*)(acc + (size_t)i * F + f) = r;      // also clears 0xAA poison
}

// ------------- edge scatter: acc[col] += dinv[row]*dinv[col]*h[row] -------
template <int F, int TPE>  // TPE threads per edge, each F/TPE feats (mult of 4)
__global__ void k_edge_scatter(const int* __restrict__ row, const int* __restrict__ col,
                               const float* __restrict__ dinv, const float* __restrict__ h,
                               float* __restrict__ acc, int E) {
  constexpr int FPT = F / TPE;
  int tid = blockIdx.x * blockDim.x + threadIdx.x;
  int e = tid / TPE;
  if (e >= E) return;
  int f0 = (tid % TPE) * FPT;
  int r = row[e], c = col[e];
  float nrm = dinv[r] * dinv[c];
  const float* hs = h   + (size_t)r * F + f0;
  float*       ds = acc + (size_t)c * F + f0;
#pragma unroll
  for (int q = 0; q < FPT; q += 4) {
    float4 hv = *(const float4*)(hs + q);
    unsafeAtomicAdd(ds + q + 0, nrm * hv.x);
    unsafeAtomicAdd(ds + q + 1, nrm * hv.y);
    unsafeAtomicAdd(ds + q + 2, nrm * hv.z);
    unsafeAtomicAdd(ds + q + 3, nrm * hv.w);
  }
}

// ------------- h1 = relu(acc1) @ W2^T + b2   (32 -> 8) --------------------
__global__ __launch_bounds__(256) void k_gemm2_relu(
    const float* __restrict__ acc1, const float* __restrict__ W2,
    const float* __restrict__ b2, float* __restrict__ h1, int N) {
  __shared__ float sW2[8 * 32];
  __shared__ float sb2[8];
  for (int t = threadIdx.x; t < 256; t += blockDim.x) sW2[t] = W2[t];
  if (threadIdx.x < 8) sb2[threadIdx.x] = b2[threadIdx.x];
  __syncthreads();
  int i = blockIdx.x * blockDim.x + threadIdx.x;
  if (i >= N) return;
  float v[32];
  const float4* p = (const float4*)(acc1 + (size_t)i * 32);
#pragma unroll
  for (int q = 0; q < 8; ++q) {
    float4 t = p[q];
    v[4*q+0] = fmaxf(t.x, 0.f); v[4*q+1] = fmaxf(t.y, 0.f);
    v[4*q+2] = fmaxf(t.z, 0.f); v[4*q+3] = fmaxf(t.w, 0.f);
  }
#pragma unroll
  for (int o = 0; o < 8; ++o) {
    float s = sb2[o];
#pragma unroll
    for (int k = 0; k < 32; ++k) s = fmaf(v[k], sW2[o * 32 + k], s);
    h1[(size_t)i * 8 + o] = s;
  }
}

// ------------- log_softmax + Linear(9,1) + relu ---------------------------
__global__ void k_final(const float* __restrict__ acc2, const float* __restrict__ x1,
                        const float* __restrict__ W3, const float* __restrict__ b3,
                        float* __restrict__ out, float* __restrict__ emb, int N) {
  int i = blockIdx.x * blockDim.x + threadIdx.x;
  if (i >= N) return;
  float v[8];
  const float4* p = (const float4*)(acc2 + (size_t)i * 8);
  float4 t0 = p[0], t1 = p[1];
  v[0]=t0.x; v[1]=t0.y; v[2]=t0.z; v[3]=t0.w; v[4]=t1.x; v[5]=t1.y; v[6]=t1.z; v[7]=t1.w;
  float m = v[0];
#pragma unroll
  for (int j = 1; j < 8; ++j) m = fmaxf(m, v[j]);
  float s = 0.f;
#pragma unroll
  for (int j = 0; j < 8; ++j) s += __expf(v[j] - m);
  float lse = m + __logf(s);
  float o = b3[0];
#pragma unroll
  for (int j = 0; j < 8; ++j) {
    float e = v[j] - lse;
    emb[(size_t)i * 8 + j] = e;
    o = fmaf(e, W3[j], o);
  }
  o = fmaf(x1[i], W3[8], o);
  out[i] = fmaxf(o, 0.f);
}

extern "C" void kernel_launch(void* const* d_in, const int* in_sizes, int n_in,
                              void* d_out, int out_size, void* d_ws, size_t ws_size,
                              hipStream_t stream) {
  const float* x  = (const float*)d_in[0];
  const int*   ei = (const int*)d_in[1];   // int32 per harness convention
  const float* x1 = (const float*)d_in[2];
  const float* W1 = (const float*)d_in[3];
  const float* b1 = (const float*)d_in[4];
  const float* W2 = (const float*)d_in[5];
  const float* b2 = (const float*)d_in[6];
  const float* W3 = (const float*)d_in[7];
  const float* b3 = (const float*)d_in[8];

  const int N = in_sizes[0] / 128;
  const int E = in_sizes[1] / 2;
  const int* row = ei;
  const int* col = ei + E;

  float* out = (float*)d_out;   // [N]
  float* emb = out + N;         // [N,8]

  // workspace carve-out: N*(1+32+32+8+8) floats ~= 32.4 MB
  float* dinv = (float*)d_ws;
  float* h0   = dinv + N;
  float* acc1 = h0   + (size_t)N * 32;
  float* h1   = acc1 + (size_t)N * 32;
  float* acc2 = h1   + (size_t)N * 8;

  const int B = 256;
  auto g = [](long long n, int b) { return (int)((n + b - 1) / b); };

  // degrees over row (incl. self loops), then dinv = deg^-1/2
  k_deg_init <<<g(N, B), B, 0, stream>>>(dinv, N);
  k_deg_count<<<g(E, B), B, 0, stream>>>(row, dinv, E);
  k_deg_rsqrt<<<g(N, B), B, 0, stream>>>(dinv, N);

  // conv1 linear on the matrix pipe: one wave per 16-row stripe (both col tiles)
  const int stripes = (N + 15) / 16;
  k_gemm1_wmma<<<g((long long)stripes * 32, 256), 256, 0, stream>>>(x, W1, b1, h0, N);

  // conv1 aggregate: self loops then edge scatter (fp32 HW atomics)
  k_self_loop<32><<<g((long long)N * 8, B), B, 0, stream>>>(h0, dinv, acc1, N);
  k_edge_scatter<32, 2><<<g((long long)E * 2, B), B, 0, stream>>>(row, col, dinv, h0, acc1, E);

  // relu + conv2 linear (32 -> 8)
  k_gemm2_relu<<<g(N, B), B, 0, stream>>>(acc1, W2, b2, h1, N);

  // conv2 aggregate
  k_self_loop<8><<<g((long long)N * 2, B), B, 0, stream>>>(h1, dinv, acc2, N);
  k_edge_scatter<8, 2><<<g((long long)E * 2, B), B, 0, stream>>>(row, col, dinv, h1, acc2, E);

  // log_softmax + head
  k_final<<<g(N, B), B, 0, stream>>>(acc2, x1, W3, b3, out, emb, N);
}
